// Encoder_34170759807509
// MI455X (gfx1250) — matrix operand order repaired and checked by reading, not compiled, not run on hardware
//
#include <hip/hip_runtime.h>
#include <hip/hip_bf16.h>

typedef __bf16 bf16_t;
typedef __attribute__((ext_vector_type(16))) __bf16 v16bf;
typedef __attribute__((ext_vector_type(8)))  __bf16 v8bf;
typedef __attribute__((ext_vector_type(8)))  float   v8f;
typedef __attribute__((ext_vector_type(4))) unsigned int u32x4;
typedef __attribute__((ext_vector_type(8))) int          i32x8;
typedef __attribute__((ext_vector_type(4))) int          i32x4;

#define B_   32
#define S_   256
#define E_   512
#define U_   512
#define NG_  1536          // 3*U
#define SB_  (S_ * B_)     // 8192 rows (s*32 + b)
#define WELEMS (NG_ * U_)  // one transposed weight matrix, 786432 elems

// scan-kernel LDS layout (dynamic LDS, assumed base offset 0)
#define LDS_H_OFF      0                        // 2 x 32*512 bf16 = 64 KB
#define LDS_MX_OFF     65536                    // 2 x 96 KB mx tiles
#define MX_TILE_BYTES  (B_ * NG_ * 2)           // 98304
#define LDS_PREV_OFF   (LDS_MX_OFF + 2 * MX_TILE_BYTES)   // 32 KB bf16
#define SCAN_LDS_BYTES (LDS_PREV_OFF + B_ * U_ * 2)       // 294912

// ---- WMMA fragment helpers (layouts per CDNA5 ISA 7.12.2) ----

// A (16x32 bf16): lane m = lane&15, half = lane>>4.
// elems 0..7  -> K = K0 + half*8 + e ; elems 8..15 -> K = K0 + 16 + half*8 + (e-8)
__device__ __forceinline__ v16bf frag_a(const bf16_t* row, int K0, int half) {
  v8bf lo = *(const v8bf*)(row + K0 + half * 8);
  v8bf hi = *(const v8bf*)(row + K0 + 16 + half * 8);
  return __builtin_shufflevector(lo, hi, 0,1,2,3,4,5,6,7,8,9,10,11,12,13,14,15);
}

// B (32x16 bf16), weights stored transposed (N,K): lane n = lane&15, half = lane>>4.
// elems 0..7 -> K = K0 + half*8 + e ; elems 8..15 -> K = K0 + 16 + half*8 + (e-8)
__device__ __forceinline__ v16bf frag_b(const bf16_t* row, int K0, int half) {
  v8bf lo = *(const v8bf*)(row + K0 + half * 8);
  v8bf hi = *(const v8bf*)(row + K0 + 16 + half * 8);
  return __builtin_shufflevector(lo, hi, 0,1,2,3,4,5,6,7,8,9,10,11,12,13,14,15);
}

__device__ __forceinline__ v8f wmma_bf16(v16bf a, v16bf b, v8f c) {
  return __builtin_amdgcn_wmma_f32_16x16x32_bf16(false, a, false, b, (short)0, c,
                                                 false, false);
}

__device__ __forceinline__ float sigmoidf_(float x) {
  return 1.f / (1.f + __expf(-x));
}

// ---- TDM: DMA one contiguous 96 KB mx tile (32 rows x 1536 bf16) into LDS ----
// Descriptor per CDNA5 ISA:
// 1-D tile of 12288 elements of 8 bytes; tracked on TENSORcnt.
__device__ __forceinline__ void tdm_load_mx(const bf16_t* gsrc, unsigned lds_off) {
  unsigned long long ga = (unsigned long long)(uintptr_t)gsrc;
  const unsigned n8 = MX_TILE_BYTES / 8;                  // 12288
  u32x4 g0;
  g0[0] = 1u;                                             // count=1, user mode
  g0[1] = lds_off;                                        // lds_addr (bytes)
  g0[2] = (unsigned)(ga & 0xFFFFFFFFu);                   // global_addr[31:0]
  g0[3] = (unsigned)((ga >> 32) & 0x01FFFFFFu)            // global_addr[56:32]
          | 0x80000000u;                                  // type = 2 ("image")
  i32x8 g1;
  g1[0] = (3 << 16);                                      // data_size=3 (8B), mask=0
  g1[1] = (int)((n8 & 0xFFFFu) << 16);                    // tensor_dim0[15:0]
  g1[2] = (int)((n8 >> 16) | (1u << 16));                 // tensor_dim0[31:16] | tensor_dim1=1
  g1[3] = (int)((n8 & 0xFFFFu) << 16);                    // tile_dim0 = n8
  g1[4] = 0;                                              // tile_dim1=0, tile_dim2=0
  g1[5] = (int)n8;                                        // tensor_dim0_stride[31:0]
  g1[6] = 0;                                              // stride0[47:32] | stride1[15:0]
  g1[7] = 0;
  i32x4 z4 = {0, 0, 0, 0};
#if __clang_major__ >= 23
  i32x8 z8 = {0, 0, 0, 0, 0, 0, 0, 0};
  __builtin_amdgcn_tensor_load_to_lds(g0, g1, z4, z4, z8, 0);
#else
  __builtin_amdgcn_tensor_load_to_lds(g0, g1, z4, z4, 0);
#endif
}

// -------------------- embedding gather + mask --------------------
__global__ void __launch_bounds__(256)
embed_kernel(const int* __restrict__ x, const float* __restrict__ table,
             bf16_t* __restrict__ emb, int* __restrict__ maskbuf) {
  unsigned gid = blockIdx.x * 256u + threadIdx.x;   // over SB_*E_ = 4194304
  int e  = gid & (E_ - 1);
  int sb = gid >> 9;                                // s*32 + b
  int b  = sb & (B_ - 1);
  int s  = sb >> 5;
  int tok = x[b * S_ + s];
  emb[gid] = (bf16_t)table[(size_t)tok * E_ + e];
  if (e == 0) maskbuf[sb] = (tok != 0) ? 1 : 0;
}

// ---------- weight convert + transpose: (512,1536)f32 -> (1536,512)bf16 ----------
__global__ void __launch_bounds__(256)
wconv_kernel(const float* w0, const float* w1, const float* w2, const float* w3,
             const float* w4, const float* w5, const float* w6, const float* w7,
             bf16_t* __restrict__ dst) {
  const float* w;
  switch (blockIdx.y) {
    case 0: w = w0; break; case 1: w = w1; break;
    case 2: w = w2; break; case 3: w = w3; break;
    case 4: w = w4; break; case 5: w = w5; break;
    case 6: w = w6; break; default: w = w7; break;
  }
  unsigned gid = blockIdx.x * 256u + threadIdx.x;   // over WELEMS
  int k = gid & (U_ - 1);
  int n = gid >> 9;
  dst[(size_t)blockIdx.y * WELEMS + gid] = (bf16_t)w[(size_t)k * NG_ + n];
}

// -------- big parallel GEMM: (8192,512)bf16 x (512,1536) + bias -> bf16 ----------
__global__ void __launch_bounds__(256)
gemm_kernel(const bf16_t* __restrict__ A, const bf16_t* __restrict__ BT,
            const float* __restrict__ bias, bf16_t* __restrict__ C) {
  int lane = threadIdx.x & 31, wv = threadIdx.x >> 5;
  int half = lane >> 4, ln = lane & 15;
  int tile  = blockIdx.x * 8 + wv;     // 512 * 96 tiles
  int mtile = tile & 511;
  int ntile = tile >> 9;
  const bf16_t* arow = A + (size_t)(mtile * 16 + ln) * U_;
  int ncol = ntile * 16 + ln;
  const bf16_t* brow = BT + (size_t)ncol * U_;
  v8f acc = {0.f, 0.f, 0.f, 0.f, 0.f, 0.f, 0.f, 0.f};
#pragma unroll
  for (int k = 0; k < 16; ++k) {
    int K0 = k * 32;
    v16bf a = frag_a(arow, K0, half);
    v16bf b = frag_b(brow, K0, half);
    acc = wmma_bf16(a, b, acc);
  }
  float bv = bias[ncol];
#pragma unroll
  for (int r = 0; r < 8; ++r) {
    int m = mtile * 16 + half * 8 + r;           // C row (m = r + 8*half)
    C[(size_t)m * NG_ + ncol] = (bf16_t)(acc[r] + bv);
  }
}

// -------------------- sequential GRU scan: 1 workgroup per direction --------------
// LDS: h double-buffer bf16 (64K) + mx double-buffer via TDM (192K) + prev bf16 (32K)
__global__ void __launch_bounds__(1024)
scan_kernel(int layer,
            const bf16_t* __restrict__ mx_all,    // [2][SB_][1536] bf16
            const bf16_t* __restrict__ wT_all,    // 8 x (1536,512) bf16
            const float*  __restrict__ state_in,  // (4,32,512)
            const float* fb0, const float* fb1, const float* bb0, const float* bb1,
            const int*    __restrict__ maskbuf,   // (S,B)
            bf16_t*       __restrict__ aseq_all,  // [2][SB_][512] bf16
            float*        __restrict__ y,         // (B,S,1024)
            float*        __restrict__ state_out) {
  extern __shared__ char smem[];
  bf16_t* hbuf    = (bf16_t*)(smem + LDS_H_OFF);         // [2][32*512]
  bf16_t* prevout = (bf16_t*)(smem + LDS_PREV_OFF);      // [32*512]

  const int d = blockIdx.x;                              // 0=fw, 1=bw
  const bf16_t* mx  = mx_all + (size_t)d * SB_ * NG_;
  const bf16_t* rkT = wT_all + (size_t)(d * 4 + layer * 2 + 1) * WELEMS;
  const float* bias2 = (d == 0) ? (layer == 0 ? fb0 : fb1)
                                : (layer == 0 ? bb0 : bb1);
  const float* rb = bias2 + NG_;                         // row 1 (h-side bias)
  bf16_t* aseq = aseq_all + (size_t)d * SB_ * U_;

  const int tid  = threadIdx.x;
  const int lane = tid & 31, wv = tid >> 5;
  const int half = lane >> 4, ln = lane & 15;

  // init: h from state, prev_out = 0
  const float* h0 = state_in + (size_t)(d * 2 + layer) * B_ * U_;
  for (int i = tid; i < B_ * U_; i += 1024) {
    hbuf[i] = (bf16_t)h0[i];
    prevout[i] = (bf16_t)0.f;
  }
  // prologue: TDM-prefetch mx tile for step 0 into buffer 0
  if (wv == 0) {
    int s0 = (d == 0) ? 0 : S_ - 1;
    tdm_load_mx(mx + (size_t)s0 * B_ * NG_, LDS_MX_OFF);
  }
  __syncthreads();

  for (int tt = 0; tt < S_; ++tt) {
    const int s = (d == 0) ? tt : (S_ - 1 - tt);
    const bf16_t* hcur = hbuf + (tt & 1) * (B_ * U_);
    bf16_t*       hnxt = hbuf + ((tt & 1) ^ 1) * (B_ * U_);

    // TDM-prefetch next step's mx tile into the other buffer (overlaps WMMA loop)
    if (wv == 0 && tt + 1 < S_) {
      int sn = (d == 0) ? (tt + 1) : (S_ - 2 - tt);
      tdm_load_mx(mx + (size_t)sn * B_ * NG_,
                  LDS_MX_OFF + ((tt + 1) & 1) * MX_TILE_BYTES);
    }

    v8f accz[2], accr[2], acch[2];
#pragma unroll
    for (int pp = 0; pp < 2; ++pp) {
      const int patch = wv + pp * 32;        // 64 patches: (utile 0..31) x (mtile 0..1)
      const int mtile = patch & 1;
      const int utile = patch >> 1;
      const int mrow  = mtile * 16 + ln;     // A row for this lane
      const int ucol  = utile * 16 + ln;     // C col / u index for this lane
      const bf16_t* arow = hcur + (size_t)mrow * U_;
      const bf16_t* brz  = rkT + (size_t)(ucol) * U_;
      const bf16_t* brr  = rkT + (size_t)(U_ + ucol) * U_;
      const bf16_t* brh  = rkT + (size_t)(2 * U_ + ucol) * U_;

      v8f az = {0.f,0.f,0.f,0.f,0.f,0.f,0.f,0.f};
      v8f ar = {0.f,0.f,0.f,0.f,0.f,0.f,0.f,0.f};
      v8f ah = {0.f,0.f,0.f,0.f,0.f,0.f,0.f,0.f};
#pragma unroll 4
      for (int k = 0; k < 16; ++k) {
        int K0 = k * 32;
        v16bf a  = frag_a(arow, K0, half);     // ds_load_b128 x2, reused 3x
        v16bf b0 = frag_b(brz, K0, half);
        v16bf b1 = frag_b(brr, K0, half);
        v16bf b2 = frag_b(brh, K0, half);
        az = wmma_bf16(a, b0, az);
        ar = wmma_bf16(a, b1, ar);
        ah = wmma_bf16(a, b2, ah);
      }
      accz[pp] = az; accr[pp] = ar; acch[pp] = ah;
    }

    // ensure this step's mx tile (issued one step earlier, in-order) has landed
    if (wv == 0) {
      if (tt + 1 < S_) __builtin_amdgcn_s_wait_tensorcnt(1);  // next-step load may fly
      else             __builtin_amdgcn_s_wait_tensorcnt(0);
    }
    __syncthreads();

    const bf16_t* mxl =
        (const bf16_t*)(smem + LDS_MX_OFF + (tt & 1) * MX_TILE_BYTES);

#pragma unroll
    for (int pp = 0; pp < 2; ++pp) {
      const int patch = wv + pp * 32;
      const int mtile = patch & 1;
      const int utile = patch >> 1;
      const int ucol  = utile * 16 + ln;
      const float rbz = rb[ucol], rbr = rb[U_ + ucol], rbh = rb[2 * U_ + ucol];
#pragma unroll
      for (int r = 0; r < 8; ++r) {
        const int m = mtile * 16 + half * 8 + r;         // C row
        const int mk = maskbuf[s * B_ + m];
        const int mrowl = m * NG_;
        float xz = (float)mxl[mrowl + ucol];
        float xr = (float)mxl[mrowl + U_ + ucol];
        float xh = (float)mxl[mrowl + 2 * U_ + ucol];
        float iz = accz[pp][r] + rbz;
        float ir = accr[pp][r] + rbr;
        float ih = acch[pp][r] + rbh;
        float hold = (float)hcur[(size_t)m * U_ + ucol];
        float z  = sigmoidf_(xz + iz);
        float rg = sigmoidf_(xr + ir);
        float hh = tanhf(xh + rg * ih);
        float hn = z * hold + (1.f - z) * hh;            // candidate new h
        float hnew = mk ? hn : hold;
        hnxt[(size_t)m * U_ + ucol] = (bf16_t)hnew;
        if (layer == 0) {
          aseq[((size_t)s * B_ + m) * U_ + ucol] = (bf16_t)hnew;
        } else {
          const size_t pidx = (size_t)m * U_ + ucol;
          float out = mk ? hn : (float)prevout[pidx];
          prevout[pidx] = (bf16_t)out;
          y[((size_t)m * S_ + s) * 1024 + d * U_ + ucol] = out;
        }
      }
    }
    __syncthreads();
  }

  // final h lives in buffer 0 after S_=256 (even) steps
  float* so = state_out + (size_t)(d * 2 + layer) * B_ * U_;
  for (int i = tid; i < B_ * U_; i += 1024)
    so[i] = (float)hbuf[i];
}

extern "C" void kernel_launch(void* const* d_in, const int* in_sizes, int n_in,
                              void* d_out, int out_size, void* d_ws, size_t ws_size,
                              hipStream_t stream) {
  (void)in_sizes; (void)n_in; (void)out_size; (void)ws_size;
  const int*   x         = (const int*)d_in[0];
  const float* state     = (const float*)d_in[1];
  const float* emb_table = (const float*)d_in[2];
  const float* fk0 = (const float*)d_in[3],  *frk0 = (const float*)d_in[4],
             * fb0 = (const float*)d_in[5];
  const float* fk1 = (const float*)d_in[6],  *frk1 = (const float*)d_in[7],
             * fb1 = (const float*)d_in[8];
  const float* bk0 = (const float*)d_in[9],  *brk0 = (const float*)d_in[10],
             * bb0 = (const float*)d_in[11];
  const float* bk1 = (const float*)d_in[12], *brk1 = (const float*)d_in[13],
             * bb1 = (const float*)d_in[14];

  float* y         = (float*)d_out;
  float* state_out = y + (size_t)B_ * S_ * 1024;

  // workspace layout (~88 MB)
  bf16_t* ws_emb  = (bf16_t*)d_ws;                          // SB_*512 bf16
  bf16_t* ws_w    = ws_emb + (size_t)SB_ * E_;              // 8 * WELEMS bf16
  int*    ws_mask = (int*)(ws_w + (size_t)8 * WELEMS);      // SB_ ints
  bf16_t* ws_aseq = (bf16_t*)(ws_mask + SB_);               // 2 * SB_*512 bf16
  bf16_t* ws_mx   = ws_aseq + (size_t)2 * SB_ * U_;         // 2 * SB_*1536 bf16

  // 1) gather + mask
  embed_kernel<<<(SB_ * E_) / 256, 256, 0, stream>>>(x, emb_table, ws_emb, ws_mask);

  // 2) weights -> bf16 transposed; order: fk0, frk0, fk1, frk1, bk0, brk0, bk1, brk1
  wconv_kernel<<<dim3(WELEMS / 256, 8), 256, 0, stream>>>(
      fk0, frk0, fk1, frk1, bk0, brk0, bk1, brk1, ws_w);

  // 3) layer-0 input GEMMs (fully parallel)
  gemm_kernel<<<6144, 256, 0, stream>>>(ws_emb, ws_w + (size_t)0 * WELEMS, fb0, ws_mx);
  gemm_kernel<<<6144, 256, 0, stream>>>(ws_emb, ws_w + (size_t)4 * WELEMS, bb0,
                                        ws_mx + (size_t)SB_ * NG_);

  // 4) layer-0 recurrent scan (writes a-sequence)
  scan_kernel<<<2, 1024, SCAN_LDS_BYTES, stream>>>(
      0, ws_mx, ws_w, state, fb0, fb1, bb0, bb1, ws_mask, ws_aseq, y, state_out);

  // 5) layer-1 input GEMMs over the a-sequence (fully parallel, reuses mx buffer)
  gemm_kernel<<<6144, 256, 0, stream>>>(ws_aseq, ws_w + (size_t)2 * WELEMS, fb1, ws_mx);
  gemm_kernel<<<6144, 256, 0, stream>>>(ws_aseq + (size_t)SB_ * U_,
                                        ws_w + (size_t)6 * WELEMS, bb1,
                                        ws_mx + (size_t)SB_ * NG_);

  // 6) layer-1 recurrent scan (writes y with masked carry + final states)
  scan_kernel<<<2, 1024, SCAN_LDS_BYTES, stream>>>(
      1, ws_mx, ws_w, state, fb0, fb1, bb0, bb1, ws_mask, ws_aseq, y, state_out);
}
